// SparseAutoencoder_36137854828625
// MI455X (gfx1250) — matrix operand order repaired and checked by reading, not compiled, not run on hardware
//
#include <hip/hip_runtime.h>

#define B_DIM 4096
#define D_DIM 768
#define L_DIM 24576
#define K_TOP 64

typedef unsigned int u32;
typedef unsigned short u16;
typedef __attribute__((ext_vector_type(16))) __bf16 v16bf;
typedef __attribute__((ext_vector_type(8)))  float  v8f;
typedef __attribute__((ext_vector_type(8)))  unsigned int v8u;
typedef __attribute__((ext_vector_type(4)))  unsigned int u32x4;

// ---------- helpers ----------
__device__ inline u16 f32_to_bf16_rne(float f, float* hi_back) {
    u32 u = __float_as_uint(f);
    u32 r = u + 0x7FFFu + ((u >> 16) & 1u);
    u16 h = (u16)(r >> 16);
    if (hi_back) *hi_back = __uint_as_float(((u32)h) << 16);
    return h;
}

// monotone float -> uint key (larger float => larger key)
__device__ inline u32 fkey(float v) {
    u32 u = __float_as_uint(v);
    u32 m = (u32)(-(int)(u >> 31)) | 0x80000000u;
    return u ^ m;
}

__device__ inline v16bf make_frag(const u32x4* lds, int i0, int i1) {
    u32x4 a = lds[i0];
    u32x4 b = lds[i1];
    v8u u = {a.x, a.y, a.z, a.w, b.x, b.y, b.z, b.w};
    return __builtin_bit_cast(v16bf, u);
}

// ---------- 1) preprocess: shift, center, L2-normalize; emit bf16 hi/lo ----------
__global__ __launch_bounds__(256) void prep_kernel(const float* __restrict__ x,
                                                   const float* __restrict__ bpre,
                                                   u16* __restrict__ xhi,
                                                   u16* __restrict__ xlo,
                                                   float* __restrict__ meanArr,
                                                   float* __restrict__ normArr) {
    const int b = blockIdx.x, tid = threadIdx.x;
    __shared__ float red[256];
    float s[3];
    for (int j = 0; j < 3; ++j) {
        int d = tid + j * 256;
        s[j] = x[b * D_DIM + d] - bpre[d];
    }
    float loc = s[0] + s[1] + s[2];
    red[tid] = loc;
    __syncthreads();
    for (int off = 128; off > 0; off >>= 1) {
        if (tid < off) red[tid] += red[tid + off];
        __syncthreads();
    }
    float mean = red[0] * (1.0f / (float)D_DIM);
    __syncthreads();
    float ss = 0.f;
    for (int j = 0; j < 3; ++j) { s[j] -= mean; ss += s[j] * s[j]; }
    red[tid] = ss;
    __syncthreads();
    for (int off = 128; off > 0; off >>= 1) {
        if (tid < off) red[tid] += red[tid + off];
        __syncthreads();
    }
    float nrm = fmaxf(sqrtf(red[0]), 1e-8f);
    for (int j = 0; j < 3; ++j) {
        int d = tid + j * 256;
        float f = s[j] / nrm;
        float hf;
        u16 h = f32_to_bf16_rne(f, &hf);
        u16 l = f32_to_bf16_rne(f - hf, nullptr);
        xhi[b * D_DIM + d] = h;
        xlo[b * D_DIM + d] = l;
    }
    if (tid == 0) { meanArr[b] = mean; normArr[b] = nrm; }
}

// ---------- 2) split W_enc into bf16 hi/lo ----------
__global__ __launch_bounds__(256) void wsplit_kernel(const float* __restrict__ W,
                                                     u16* __restrict__ whi,
                                                     u16* __restrict__ wlo) {
    const int base = blockIdx.x * 2048 + threadIdx.x;
    for (int j = 0; j < 8; ++j) {
        int i = base + j * 256;
        float f = W[i];
        float hf;
        u16 h = f32_to_bf16_rne(f, &hf);
        u16 l = f32_to_bf16_rne(f - hf, nullptr);
        whi[i] = h;
        wlo[i] = l;
    }
}

// ---------- 3) encoder GEMM: pre = x_norm @ W_enc^T via split-bf16 WMMA ----------
// WG: 256 thr (8 waves), tile 128M x 128N, K-step 32, LDS-staged.
__global__ __launch_bounds__(256) void enc_gemm(const u16* __restrict__ xhi,
                                                const u16* __restrict__ xlo,
                                                const u16* __restrict__ whi,
                                                const u16* __restrict__ wlo,
                                                float* __restrict__ pre) {
    __shared__ u32x4 sAh[512], sAl[512], sBh[512], sBl[512];
    const int tid = threadIdx.x;
    const int lane = tid & 31, wave = tid >> 5;
    const int wm = (wave >> 1) * 32;   // wave M offset in WG tile (4 rows of waves)
    const int wn = (wave & 1) * 64;    // wave N offset (2 cols of waves)
    const int lr = lane & 15, ks = lane >> 4;
    const int mBase = blockIdx.y * 128;
    const int nBase = blockIdx.x * 128;
    const u32x4* xh4 = (const u32x4*)xhi;  // row = 96 u32x4
    const u32x4* xl4 = (const u32x4*)xlo;
    const u32x4* wh4 = (const u32x4*)whi;
    const u32x4* wl4 = (const u32x4*)wlo;

    v8f acc[2][4];
    for (int mt = 0; mt < 2; ++mt)
        for (int nt = 0; nt < 4; ++nt)
            acc[mt][nt] = (v8f)0.f;

    for (int k0 = 0; k0 < D_DIM; k0 += 32) {
        const int kq = k0 >> 3;  // u32x4 offset within a row (32 halves = 4 u32x4)
        // cooperative stage: 128 rows x 4 u32x4 per matrix
        for (int i = 0; i < 2; ++i) {
            int q = tid + i * 256;          // 0..511
            int r = q >> 2, c = q & 3;
            sAh[q] = xh4[(mBase + r) * 96 + kq + c];
            sAl[q] = xl4[(mBase + r) * 96 + kq + c];
            sBh[q] = wh4[(nBase + r) * 96 + kq + c];
            sBl[q] = wl4[(nBase + r) * 96 + kq + c];
        }
        __syncthreads();

        // A fragment (16x32 bf16): lane<16 -> K0..7,K16..23 ; lane>=16 -> K8..15,K24..31
        v16bf Ah[2], Al[2], Bh[4], Bl[4];
        for (int mt = 0; mt < 2; ++mt) {
            int r4 = (wm + mt * 16 + lr) * 4;
            Ah[mt] = make_frag(sAh, r4 + ks, r4 + 2 + ks);
            Al[mt] = make_frag(sAl, r4 + ks, r4 + 2 + ks);
        }
        // B fragment (32x16 bf16): lane<16 -> K0..7,K16..23 ; lane>=16 -> K8..15,K24..31 (N = lane&15)
        for (int nt = 0; nt < 4; ++nt) {
            int r4 = (wn + nt * 16 + lr) * 4;
            Bh[nt] = make_frag(sBh, r4 + ks, r4 + 2 + ks);
            Bl[nt] = make_frag(sBl, r4 + ks, r4 + 2 + ks);
        }
        // split-precision: hi*hi + hi*lo + lo*hi
        for (int mt = 0; mt < 2; ++mt)
            for (int nt = 0; nt < 4; ++nt) {
                acc[mt][nt] = __builtin_amdgcn_wmma_f32_16x16x32_bf16(
                    false, Ah[mt], false, Bh[nt], (short)0, acc[mt][nt], false, false);
                acc[mt][nt] = __builtin_amdgcn_wmma_f32_16x16x32_bf16(
                    false, Ah[mt], false, Bl[nt], (short)0, acc[mt][nt], false, false);
                acc[mt][nt] = __builtin_amdgcn_wmma_f32_16x16x32_bf16(
                    false, Al[mt], false, Bh[nt], (short)0, acc[mt][nt], false, false);
            }
        __syncthreads();
    }

    // epilogue: C layout VGPR j -> M = j + 8*ks, N = lane&15
    for (int mt = 0; mt < 2; ++mt)
        for (int nt = 0; nt < 4; ++nt) {
            int grow0 = mBase + wm + mt * 16 + ks * 8;
            int gcol  = nBase + wn + nt * 16 + lr;
            for (int j = 0; j < 8; ++j)
                pre[(grow0 + j) * L_DIM + gcol] = acc[mt][nt][j];
        }
}

// ---------- 4) top-64 per row via 4-pass radix select; masked ReLU in place ----------
__global__ __launch_bounds__(256) void topk_kernel(float* __restrict__ z,
                                                   int* __restrict__ tkIdx,
                                                   float* __restrict__ tkVal) {
    const int b = blockIdx.x, tid = threadIdx.x;
    float* rowp = z + (size_t)b * L_DIM;
    __shared__ u32 hist[256];
    __shared__ u32 shPrefix, shRem, eqTaken, selCnt;
    if (tid == 0) { shPrefix = 0u; shRem = K_TOP; eqTaken = 0u; selCnt = 0u; }
    __syncthreads();

    for (int p = 3; p >= 0; --p) {
        hist[tid] = 0u;
        __syncthreads();
        const u32 prefix = shPrefix;
        const u32 pmask = (p == 3) ? 0u : (0xFFFFFFFFu << (8 * (p + 1)));
        for (int j = 0; j < 96; ++j) {
            int l = tid + j * 256;
            u32 k = fkey(rowp[l]);
            if ((k & pmask) == prefix) atomicAdd(&hist[(k >> (8 * p)) & 255u], 1u);
        }
        __syncthreads();
        if (tid == 0) {
            u32 rem = shRem, cum = 0u;
            for (int bk = 255; bk >= 0; --bk) {
                u32 h = hist[bk];
                if (cum + h >= rem) {
                    shPrefix = prefix | ((u32)bk << (8 * p));
                    shRem = rem - cum;
                    break;
                }
                cum += h;
            }
        }
        __syncthreads();
    }

    const u32 T = shPrefix;
    const u32 numEq = shRem;
    for (int j = 0; j < 96; ++j) {
        int l = tid + j * 256;
        float v = rowp[l];
        u32 k = fkey(v);
        bool sel = false;
        if (k > T) sel = true;
        else if (k == T) { u32 pos = atomicAdd(&eqTaken, 1u); sel = (pos < numEq); }
        float zv = sel ? fmaxf(v, 0.f) : 0.f;
        rowp[l] = zv;
        if (sel) {
            u32 s = atomicAdd(&selCnt, 1u);
            if (s < K_TOP) { tkIdx[b * K_TOP + s] = l; tkVal[b * K_TOP + s] = zv; }
        }
    }
}

// ---------- 5) sparse decoder + restore ----------
__global__ __launch_bounds__(256) void dec_kernel(const int* __restrict__ tkIdx,
                                                  const float* __restrict__ tkVal,
                                                  const float* __restrict__ Wdec,
                                                  const float* __restrict__ bpre,
                                                  const float* __restrict__ meanArr,
                                                  const float* __restrict__ normArr,
                                                  float* __restrict__ xhat) {
    const int b = blockIdx.x, tid = threadIdx.x;
    __shared__ float sv[K_TOP];
    __shared__ int si[K_TOP];
    if (tid < K_TOP) { sv[tid] = tkVal[b * K_TOP + tid]; si[tid] = tkIdx[b * K_TOP + tid]; }
    __syncthreads();
    const float mean = meanArr[b], nrm = normArr[b];
    for (int j = 0; j < 3; ++j) {
        int d = tid + j * 256;
        float acc = 0.f;
#pragma unroll
        for (int k = 0; k < K_TOP; ++k)
            acc = fmaf(sv[k], Wdec[d * L_DIM + si[k]], acc);
        xhat[b * D_DIM + d] = acc * nrm + mean + bpre[d];
    }
}

// ---------- launch ----------
extern "C" void kernel_launch(void* const* d_in, const int* in_sizes, int n_in,
                              void* d_out, int out_size, void* d_ws, size_t ws_size,
                              hipStream_t stream) {
    (void)in_sizes; (void)n_in; (void)out_size; (void)ws_size;
    const float* x    = (const float*)d_in[0];
    const float* Wenc = (const float*)d_in[1];
    const float* Wdec = (const float*)d_in[2];
    const float* bpre = (const float*)d_in[3];

    float* xhat = (float*)d_out;
    float* z    = xhat + (size_t)B_DIM * D_DIM;  // z region; holds raw pre, then masked z

    char* w = (char*)d_ws;
    u16* whi = (u16*)w;               w += (size_t)L_DIM * D_DIM * sizeof(u16);
    u16* wlo = (u16*)w;               w += (size_t)L_DIM * D_DIM * sizeof(u16);
    u16* xhi = (u16*)w;               w += (size_t)B_DIM * D_DIM * sizeof(u16);
    u16* xlo = (u16*)w;               w += (size_t)B_DIM * D_DIM * sizeof(u16);
    float* meanArr = (float*)w;       w += (size_t)B_DIM * sizeof(float);
    float* normArr = (float*)w;       w += (size_t)B_DIM * sizeof(float);
    float* tkVal   = (float*)w;       w += (size_t)B_DIM * K_TOP * sizeof(float);
    int*   tkIdx   = (int*)w;

    prep_kernel<<<B_DIM, 256, 0, stream>>>(x, bpre, xhi, xlo, meanArr, normArr);
    wsplit_kernel<<<(L_DIM * D_DIM) / 2048, 256, 0, stream>>>(Wenc, whi, wlo);
    dim3 ggrid(L_DIM / 128, B_DIM / 128);
    enc_gemm<<<ggrid, 256, 0, stream>>>(xhi, xlo, whi, wlo, z);
    topk_kernel<<<B_DIM, 256, 0, stream>>>(z, tkIdx, tkVal);
    dec_kernel<<<B_DIM, 256, 0, stream>>>(tkIdx, tkVal, Wdec, bpre, meanArr, normArr, xhat);
}
